// MultiHeadAttention_7421703487593
// MI455X (gfx1250) — matrix operand order repaired and checked by reading, not compiled, not run on hardware
//
#include <hip/hip_runtime.h>

// ---------- types ----------
typedef _Float16 h4_t  __attribute__((ext_vector_type(4)));
typedef _Float16 h8_t  __attribute__((ext_vector_type(8)));
typedef _Float16 h16_t __attribute__((ext_vector_type(16)));
typedef float    f8_t  __attribute__((ext_vector_type(8)));
typedef int      v4i_t __attribute__((ext_vector_type(4)));

union H16 { h16_t v; h8_t h[2]; };

#define S_LEN   2048
#define D_MODEL 1024
#define NHEADS  16
#define HEADD   64

// ---------- helpers ----------
__device__ __forceinline__ f8_t wmma_f16(h16_t a, h16_t b, f8_t c) {
  // v_wmma_f32_16x16x32_f16 : D = A(16x32 f16) * B(32x16 f16) + C(16x16 f32)
  return __builtin_amdgcn_wmma_f32_16x16x32_f16(false, a, false, b, (short)0, c,
                                                false, false);
}

// async global->LDS copy of one 16B segment per lane (ASYNCcnt path)
__device__ __forceinline__ void async_g2l_b128(const _Float16* g, _Float16* l) {
#if __has_builtin(__builtin_amdgcn_global_load_async_to_lds_b128)
  __builtin_amdgcn_global_load_async_to_lds_b128(
      (__attribute__((address_space(1))) v4i_t*)g,
      (__attribute__((address_space(3))) v4i_t*)l, 0, 0);
#else
  unsigned loff = (unsigned)(size_t)(__attribute__((address_space(3))) void*)l;
  unsigned long long ga = (unsigned long long)g;
  asm volatile("global_load_async_to_lds_b128 %0, %1, off"
               :: "v"(loff), "v"(ga)
               : "memory");
#endif
}

__device__ __forceinline__ void wait_async0() {
#if __has_builtin(__builtin_amdgcn_s_wait_asynccnt)
  __builtin_amdgcn_s_wait_asynccnt(0);
#else
  asm volatile("s_wait_asynccnt 0x0" ::: "memory");
#endif
}

// XOR lane shuffle via ds_swizzle (group-of-32 mode: xor in offset[14:10], and=0x1F)
template <int MASK>
__device__ __forceinline__ float swz_xor(float v) {
  int i = __builtin_amdgcn_ds_swizzle(__float_as_int(v), (MASK << 10) | 0x1F);
  return __int_as_float(i);
}
__device__ __forceinline__ float half_max(float v) {   // reduce over 16-lane half
  v = fmaxf(v, swz_xor<1>(v)); v = fmaxf(v, swz_xor<2>(v));
  v = fmaxf(v, swz_xor<4>(v)); v = fmaxf(v, swz_xor<8>(v));
  return v;
}
__device__ __forceinline__ float half_sum(float v) {
  v += swz_xor<1>(v); v += swz_xor<2>(v);
  v += swz_xor<4>(v); v += swz_xor<8>(v);
  return v;
}

// ---------- kernel 1: fp32 -> f16 ----------
__global__ void mha_cvt_f16(const float* __restrict__ in, _Float16* __restrict__ out,
                            int n4) {
  int i = blockIdx.x * blockDim.x + threadIdx.x;
  if (i < n4) {
    float4 v = ((const float4*)in)[i];
    h4_t o;
    o[0] = (_Float16)v.x; o[1] = (_Float16)v.y;
    o[2] = (_Float16)v.z; o[3] = (_Float16)v.w;
    ((h4_t*)out)[i] = o;
  }
}

// ---------- kernel 2: flash attention ----------
// grid: (S/128, NHEADS, N)  block: 256 (8 waves, 16 query rows per wave)
__global__ __launch_bounds__(256, 1)
void mha_flash(const _Float16* __restrict__ qh, const _Float16* __restrict__ kh,
               const _Float16* __restrict__ vh, _Float16* __restrict__ oh) {
  // double-buffered K chunk: [32 keys][64 d], row pitch 72 halves (16B-aligned frags)
  __shared__ _Float16 Kt[2][32 * 72];
  // double-buffered V chunk transposed: [64 d][32 keys], row pitch 40 halves
  __shared__ _Float16 Vt[2][64 * 40];
  // per-wave P tile: [16 rows][32 keys], row pitch 40 halves
  __shared__ _Float16 Pb[8][16 * 40];

  const int tid  = threadIdx.x;
  const int lane = tid & 31;
  const int wv   = tid >> 5;
  const int r15  = lane & 15;
  const int hi   = lane >> 4;          // half-wave select
  const int ka8  = hi ? 8 : 0;         // unused (A fragments also use kb16)
  const int kb16 = hi ? 16 : 0;        // B-fragment K offset

  const int h = blockIdx.y;
  const int n = blockIdx.z;
  const int q0 = blockIdx.x * 128 + wv * 16;
  const size_t headbase = ((size_t)n * S_LEN) * D_MODEL + (size_t)h * HEADD;

  // --- Q tile (16 x 64) held as two A-fragments (K = d 0..31, 32..63) ---
  const _Float16* qp = qh + headbase + (size_t)(q0 + r15) * D_MODEL;
  H16 a0u, a1u;
  a0u.h[0] = *(const h8_t*)(qp + kb16);
  a0u.h[1] = *(const h8_t*)(qp + kb16 + 8);
  a1u.h[0] = *(const h8_t*)(qp + 32 + kb16);
  a1u.h[1] = *(const h8_t*)(qp + 32 + kb16 + 8);
  const h16_t aq0 = a0u.v, aq1 = a1u.v;

  float m_i[8], l_i[8];
  f8_t oacc[4] = {{}, {}, {}, {}};
#pragma unroll
  for (int r = 0; r < 8; ++r) { m_i[r] = -1e30f; l_i[r] = 0.f; }

  const int crow = tid >> 3, cseg = tid & 7;   // cooperative-load mapping

  // --- software pipeline: stage chunk 0 ---
  async_g2l_b128(kh + headbase + (size_t)crow * D_MODEL + cseg * 8,
                 &Kt[0][crow * 72 + cseg * 8]);
  h8_t vpre = *(const h8_t*)(vh + headbase + (size_t)crow * D_MODEL + cseg * 8);
  wait_async0();
#pragma unroll
  for (int j = 0; j < 8; ++j) Vt[0][(cseg * 8 + j) * 40 + crow] = vpre[j];
  __syncthreads();

  for (int kb = 0; kb < S_LEN; kb += 32) {
    const int cur = (kb >> 5) & 1;
    const int nxt = cur ^ 1;
    const bool more = (kb + 32) < S_LEN;

    // prefetch next chunk: K via async DMA (no VGPRs), V into registers
    if (more) {
      async_g2l_b128(kh + headbase + (size_t)(kb + 32 + crow) * D_MODEL + cseg * 8,
                     &Kt[nxt][crow * 72 + cseg * 8]);
      vpre = *(const h8_t*)(vh + headbase + (size_t)(kb + 32 + crow) * D_MODEL +
                            cseg * 8);
    }

    // --- scores: S[16 x 32] = Q(16x64) . K^T, two 16x16 C-tiles ---
    const _Float16* Kc = Kt[cur];
    f8_t s0 = {}, s1 = {};
    {
      H16 b;
      int key = r15;                               // tile 0: keys 0..15
      b.h[0] = *(const h8_t*)&Kc[key * 72 + kb16];
      b.h[1] = *(const h8_t*)&Kc[key * 72 + kb16 + 8];
      s0 = wmma_f16(aq0, b.v, s0);
      b.h[0] = *(const h8_t*)&Kc[key * 72 + 32 + kb16];
      b.h[1] = *(const h8_t*)&Kc[key * 72 + 32 + kb16 + 8];
      s0 = wmma_f16(aq1, b.v, s0);
      key = 16 + r15;                              // tile 1: keys 16..31
      b.h[0] = *(const h8_t*)&Kc[key * 72 + kb16];
      b.h[1] = *(const h8_t*)&Kc[key * 72 + kb16 + 8];
      s1 = wmma_f16(aq0, b.v, s1);
      b.h[0] = *(const h8_t*)&Kc[key * 72 + 32 + kb16];
      b.h[1] = *(const h8_t*)&Kc[key * 72 + 32 + kb16 + 8];
      s1 = wmma_f16(aq1, b.v, s1);
    }

    // --- online softmax (row = r + hi*8, cols across 16 lanes of the half) ---
    _Float16* pb = Pb[wv];
#pragma unroll
    for (int r = 0; r < 8; ++r) {
      float cm = half_max(fmaxf(s0[r], s1[r]));
      float mn = fmaxf(m_i[r], cm);
      float sc = __expf(m_i[r] - mn);
      float p0 = __expf(s0[r] - mn);
      float p1 = __expf(s1[r] - mn);
      float rs = half_sum(p0 + p1);
      l_i[r] = l_i[r] * sc + rs;
#pragma unroll
      for (int t = 0; t < 4; ++t) oacc[t][r] *= sc;
      m_i[r] = mn;
      int row = r + (hi ? 8 : 0);
      pb[row * 40 + r15]      = (_Float16)p0;
      pb[row * 40 + 16 + r15] = (_Float16)p1;
    }

    // --- C-layout -> A-layout via per-wave LDS tile (in-wave DS ordering) ---
    H16 pf;
    pf.h[0] = *(const h8_t*)&pb[r15 * 40 + kb16];
    pf.h[1] = *(const h8_t*)&pb[r15 * 40 + kb16 + 8];

    // --- O += P(16x32) . V(32x64), four 16x16 d-tiles ---
    const _Float16* Vc = Vt[cur];
#pragma unroll
    for (int t = 0; t < 4; ++t) {
      H16 bv;
      int drow = t * 16 + r15;
      bv.h[0] = *(const h8_t*)&Vc[drow * 40 + kb16];
      bv.h[1] = *(const h8_t*)&Vc[drow * 40 + kb16 + 8];
      oacc[t] = wmma_f16(pf.v, bv.v, oacc[t]);
    }

    // retire the async K prefetch, commit transposed V, flip buffers
    if (more) {
      wait_async0();
#pragma unroll
      for (int j = 0; j < 8; ++j) Vt[nxt][(cseg * 8 + j) * 40 + crow] = vpre[j];
    }
    __syncthreads();
  }

  // --- normalize and write f16 attention output (N,S,D layout) ---
#pragma unroll
  for (int r = 0; r < 8; ++r) {
    float inv = 1.0f / l_i[r];
    int row = q0 + r + (hi ? 8 : 0);
#pragma unroll
    for (int t = 0; t < 4; ++t) {
      oh[headbase + (size_t)row * D_MODEL + t * 16 + r15] =
          (_Float16)(oacc[t][r] * inv);
    }
  }
}

// ---------- kernel 3: output projection  out = X @ W^T + b ----------
// grid: (NOUT/64, M/128)  block: 256 (8 waves; wave = 16 rows x 64 cols)
__global__ __launch_bounds__(256, 1)
void mha_proj(const _Float16* __restrict__ xh, const _Float16* __restrict__ wh,
              const float* __restrict__ bias, float* __restrict__ out) {
  constexpr int K = D_MODEL, NOUT = D_MODEL;
  __shared__ _Float16 Wt[2][64 * 40];   // double-buffered [64 n][32 k], pitch 40

  const int tid = threadIdx.x;
  const int lane = tid & 31, wv = tid >> 5;
  const int r15 = lane & 15, hi = lane >> 4;
  const int ka8 = hi ? 8 : 0, kb16 = hi ? 16 : 0;
  const int m0 = blockIdx.y * 128 + wv * 16;
  const int n0 = blockIdx.x * 64;

  f8_t c[4] = {{}, {}, {}, {}};
  const int wrow = tid >> 2, wseg = tid & 3;
  const _Float16* xp = xh + (size_t)(m0 + r15) * K;
  const _Float16* wp = wh + (size_t)(n0 + wrow) * K + wseg * 8;

  // pipeline prologue: stage k-chunk 0
  async_g2l_b128(wp, &Wt[0][wrow * 40 + wseg * 8]);
  wait_async0();
  __syncthreads();

  for (int kc = 0; kc < K; kc += 32) {
    const int cur = (kc >> 5) & 1;
    const int nxt = cur ^ 1;
    const bool more = (kc + 32) < K;
    if (more) async_g2l_b128(wp + kc + 32, &Wt[nxt][wrow * 40 + wseg * 8]);

    H16 a;
    a.h[0] = *(const h8_t*)(xp + kc + kb16);
    a.h[1] = *(const h8_t*)(xp + kc + kb16 + 8);
#pragma unroll
    for (int t = 0; t < 4; ++t) {
      H16 b;
      b.h[0] = *(const h8_t*)&Wt[cur][(t * 16 + r15) * 40 + kb16];
      b.h[1] = *(const h8_t*)&Wt[cur][(t * 16 + r15) * 40 + kb16 + 8];
      c[t] = wmma_f16(a.v, b.v, c[t]);
    }

    if (more) wait_async0();
    __syncthreads();
  }

#pragma unroll
  for (int t = 0; t < 4; ++t) {
    int nn = n0 + t * 16 + r15;
    float bv = bias[nn];
#pragma unroll
    for (int r = 0; r < 8; ++r) {
      int m = m0 + r + (hi ? 8 : 0);
      out[(size_t)m * NOUT + nn] = c[t][r] + bv;
    }
  }
}

// ---------- launcher ----------
extern "C" void kernel_launch(void* const* d_in, const int* in_sizes, int n_in,
                              void* d_out, int out_size, void* d_ws, size_t ws_size,
                              hipStream_t stream) {
  const float* q = (const float*)d_in[0];
  const float* k = (const float*)d_in[1];
  const float* v = (const float*)d_in[2];
  const float* W = (const float*)d_in[3];
  const float* b = (const float*)d_in[4];
  float* out = (float*)d_out;

  const size_t NSD = (size_t)2 * S_LEN * D_MODEL;     // 4,194,304
  const size_t WSZ = (size_t)D_MODEL * D_MODEL;       // 1,048,576

  uint8_t* ws = (uint8_t*)d_ws;
  _Float16* qh = (_Float16*)(ws);
  _Float16* kh = (_Float16*)(ws + NSD * 2);
  _Float16* vh = (_Float16*)(ws + NSD * 4);
  _Float16* oh = (_Float16*)(ws + NSD * 6);
  _Float16* wh = (_Float16*)(ws + NSD * 8);
  (void)ws_size; (void)in_sizes; (void)n_in; (void)out_size;

  // fp32 -> f16 staging
  int n4q = (int)(NSD / 4);
  int n4w = (int)(WSZ / 4);
  mha_cvt_f16<<<(n4q + 255) / 256, 256, 0, stream>>>(q, qh, n4q);
  mha_cvt_f16<<<(n4q + 255) / 256, 256, 0, stream>>>(k, kh, n4q);
  mha_cvt_f16<<<(n4q + 255) / 256, 256, 0, stream>>>(v, vh, n4q);
  mha_cvt_f16<<<(n4w + 255) / 256, 256, 0, stream>>>(W, wh, n4w);

  // flash attention
  dim3 gA(S_LEN / 128, NHEADS, 2);
  mha_flash<<<gA, 256, 0, stream>>>(qh, kh, vh, oh);

  // projection (+bias)
  dim3 gP(D_MODEL / 64, (2 * S_LEN) / 128);
  mha_proj<<<gP, 256, 0, stream>>>(oh, wh, b, out);
}
